// RVAE_1657857376569
// MI455X (gfx1250) — matrix operand order repaired and checked by reading, not compiled, not run on hardware
//
#include <hip/hip_runtime.h>
#include <cstdint>
#include <cstddef>

// ---------------------------------------------------------------------------
// RVAE for MI455X (gfx1250, wave32, WMMA). All matmuls use
// v_wmma_f32_16x16x32_f16 (f16 inputs, f32 accumulate).
//
// 16-bit WMMA fragment layouts (CDNA5 ISA 7.12.2), wave32:
//   A (16x32, MxK): lane<16 -> row M=lane, K[0..15] (code's contiguous order, not ISA table)
//                   lane>=16 -> row M=lane-16, K[16..31]
//     => two contiguous 16B loads per lane from row-major A.
//   B (32x16, KxN): lane<16 -> col N=lane, K[0..15]; lane>=16 -> K[16..31] (same order as A)
//     => one contiguous 32B load per lane from row-major W[N][K] (B = W^T).
//   C/D (16x16 f32): elem e of v8f -> row M = e + 8*(lane>=16), col N = lane&15.
//
// X (gate pre-projections) is stored fragment-tiled:
//   X[t][nTile(64)][mTile(4)][lane(32)][e(8)]  -> one aligned v8f per tile.
//
// Persistent scans:
//  * half of the recurrent weight matrix is cached in LDS (320KB/WGP budget),
//    halving per-step L2 weight traffic on the serial critical path;
//  * loop-invariant weight accesses are indexed through a laundered zero
//    offset (asm "+s") so LLVM cannot hoist 256 VGPRs of fragments out of the
//    512-step loop (previously caused scratch spills);
//  * dynamic LDS is a TYPED extern array partitioned with pointer arithmetic
//    only (no uintptr_t casts!), so address-space inference keeps LDS traffic
//    on ds_load_b128/ds_store (32-bit addressing, DScnt) instead of flat ops.
// ---------------------------------------------------------------------------

typedef _Float16 half_t;
typedef __attribute__((ext_vector_type(16))) _Float16 v16h;
typedef __attribute__((ext_vector_type(8)))  _Float16 v8h;
typedef __attribute__((ext_vector_type(8)))  float    v8f;
typedef __attribute__((ext_vector_type(4)))  unsigned int v4u;

union HFrag { v16h v; v8h h[2]; };

#define WMMA16(a, b, c) \
  __builtin_amdgcn_wmma_f32_16x16x32_f16(false, (a), false, (b), (short)0, (c), false, false)

#if __has_builtin(__builtin_amdgcn_tanhf)
__device__ __forceinline__ float tanh_(float x) { return __builtin_amdgcn_tanhf(x); }
#else
__device__ __forceinline__ float tanh_(float x) { return tanhf(x); }
#endif

__device__ __forceinline__ v8f v8f_zero() {
  v8f z = {0.f, 0.f, 0.f, 0.f, 0.f, 0.f, 0.f, 0.f};
  return z;
}
__device__ __forceinline__ float sig_(float x) { return 1.f / (1.f + __expf(-x)); }

// Launder a zero so index arithmetic defeats LICM without breaking
// address-space inference.
__device__ __forceinline__ int opaque_zero_() {
  int z = 0;
  asm volatile("" : "+s"(z));
  return z;
}

// A-fragment: rows mBase..mBase+15, K chunk starting at k0, row-major, leading dim ld.
__device__ __forceinline__ v16h load_afrag(const half_t* A, int ld, int mBase, int k0, int lane) {
  int m  = mBase + (lane & 15);
  int kb = k0 + ((lane & 16) ? 16 : 0);
  const half_t* p = A + (size_t)m * ld + kb;
  HFrag r;
  r.h[0] = *(const v8h*)p;         // K = kb .. kb+7
  r.h[1] = *(const v8h*)(p + 8);   // K = kb+8 .. kb+15
  return r.v;
}
// B-fragment from W[N][K] row-major (computes A @ W^T): cols nBase..+15, chunk k0.
__device__ __forceinline__ v16h load_bfrag(const half_t* W, int ld, int nBase, int k0, int lane) {
  int n  = nBase + (lane & 15);
  int kb = k0 + ((lane & 16) ? 16 : 0);
  return *(const v16h*)(W + (size_t)n * ld + kb);  // 16 contiguous K values
}

// ---------------------------------------------------------------------------
// Pack fp32 -> f16 with zero padding to (Rp, Cp).
// ---------------------------------------------------------------------------
__global__ void k_pack(const float* __restrict__ src, half_t* __restrict__ dst,
                       int R, int C, int Rp, int Cp) {
  int stride = gridDim.x * blockDim.x;
  for (int idx = blockIdx.x * blockDim.x + threadIdx.x; idx < Rp * Cp; idx += stride) {
    int r = idx / Cp, c = idx % Cp;
    float v = (r < R && c < C) ? src[(size_t)r * C + c] : 0.f;
    dst[idx] = (half_t)v;
  }
}

// ---------------------------------------------------------------------------
// WMMA GEMM: C = act(A[M,KP] @ W[Np,KP]^T + b0 + b1)
// Block = 256 threads (8 waves, 2x4). Block tile 64x256, wave tile 32x64.
// ---------------------------------------------------------------------------
template <int KP, bool TILED, bool EDGE, bool EXPACT>
__global__ void __launch_bounds__(256) k_gemm(
    const half_t* __restrict__ A, const half_t* __restrict__ W,
    const float* __restrict__ b0, const float* __restrict__ b1,
    float* __restrict__ C, int Np, int Nout, int ldc) {
  int lane = threadIdx.x & 31;
  int wv   = threadIdx.x >> 5;
  int mBase = blockIdx.y * 64 + (wv >> 2) * 32;
  int nBase = blockIdx.x * 256 + (wv & 3) * 64;

  int jMax = 4;
  if (EDGE) {
    if (nBase >= Np) return;              // wave-uniform
    int rem = (Np - nBase) >> 4;          // remaining valid 16-col tiles
    jMax = rem < 4 ? rem : 4;             // scalar bound
  }

  v8f acc[2][4];
#pragma unroll
  for (int i = 0; i < 2; ++i)
#pragma unroll
    for (int j = 0; j < 4; ++j) acc[i][j] = v8f_zero();

  constexpr int nChunks = KP >> 5;
#pragma unroll
  for (int c = 0; c < nChunks; ++c) {
    int k0 = c << 5;
    v16h a0 = load_afrag(A, KP, mBase, k0, lane);
    v16h a1 = load_afrag(A, KP, mBase + 16, k0, lane);
#pragma unroll
    for (int j = 0; j < 4; ++j) {
      if (EDGE && j >= jMax) continue;    // scalar branch
      v16h b = load_bfrag(W, KP, nBase + j * 16, k0, lane);
      acc[0][j] = WMMA16(a0, b, acc[0][j]);
      acc[1][j] = WMMA16(a1, b, acc[1][j]);
    }
  }

  int laneHi = (lane >> 4), nl = lane & 15;
#pragma unroll
  for (int j = 0; j < 4; ++j) {
    if (EDGE && j >= jMax) continue;
    int n = nBase + j * 16 + nl;
    float bias = 0.f;
    if (!EDGE || n < Nout) bias = (b0 ? b0[n] : 0.f) + (b1 ? b1[n] : 0.f);
#pragma unroll
    for (int i = 0; i < 2; ++i) {
      v8f v;
#pragma unroll
      for (int e = 0; e < 8; ++e) {
        float x = acc[i][j][e] + bias;
        if (EXPACT) x = __expf(x);
        v[e] = x;
      }
      if (TILED) {
        int nT = blockIdx.x * 16 + (wv & 3) * 4 + j;
        int mT = (wv >> 2) * 2 + i;
        *(v8f*)(C + (((size_t)blockIdx.y * 64 + nT) * 4 + mT) * 256 + lane * 8) = v;
      } else {
        if (EDGE && n >= Nout) continue;  // lane mask on the single edge tile
        int mB = mBase + i * 16 + 8 * laneHi;
#pragma unroll
        for (int e = 0; e < 8; ++e) C[(size_t)(mB + e) * ldc + n] = v[e];
      }
    }
  }
}

// ---------------------------------------------------------------------------
// Persistent LSTM scan. One workgroup, 1024 threads = 32 waves on one WGP.
// LDS: h state (32KB f16) + Whh rows [0,512) i.e. gates i,f cached (256KB).
// Gates g,o stream from L2 each step. 64 wmma / wave / step.
// ---------------------------------------------------------------------------
__global__ void __launch_bounds__(1024) k_lstm_scan(
    const float* __restrict__ X, const half_t* __restrict__ Whh,
    half_t* __restrict__ Hout, int T, int reverse) {
  extern __shared__ v4u dyn_smem_v4[];               // typed -> stays AS(3)
  half_t* hsh  = (half_t*)dyn_smem_v4;               // 64x256 h state
  half_t* wlds = hsh + 64 * 256;                     // Whh rows [0,512): 256KB

  int tid = threadIdx.x, lane = tid & 31, w = tid >> 5;
  int ht = w >> 1, mPair = (w & 1) * 2;
  int laneHi = lane >> 4, nl = lane & 15;

  for (int i = tid; i < 64 * 256; i += 1024) hsh[i] = (half_t)0.f;
  {  // cooperative prefill of the LDS weight cache (512 rows x 256 f16)
    const v4u* src = (const v4u*)Whh;
    v4u* dst = (v4u*)wlds;
    for (int i = tid; i < (512 * 256) / 8; i += 1024) dst[i] = src[i];
  }
  v8f c[2];
  c[0] = v8f_zero();
  c[1] = v8f_zero();
  __syncthreads();

  for (int s = 0; s < T; ++s) {
    int t = reverse ? (T - 1 - s) : s;
    int zo = opaque_zero_();             // defeats LICM of invariant fragments
    const half_t* WhhO = Whh + zo;       // stays global AS
    const half_t* wldsO = wlds + zo;     // stays LDS AS
    const float* Xt = X + (size_t)t * (64 * 1024);

    v8f acc[4][2];
#pragma unroll
    for (int g = 0; g < 4; ++g)
#pragma unroll
      for (int mi = 0; mi < 2; ++mi)
        acc[g][mi] =
            *(const v8f*)(Xt + (((g * 16 + ht) * 4) + (mPair + mi)) * 256 + lane * 8);

    if (s + 1 < T) {  // prefetch next step's 256KB X block (overlaps the wmma work)
      const float* Xn = X + (size_t)(reverse ? t - 1 : t + 1) * (64 * 1024);
      __builtin_prefetch(Xn + (size_t)tid * 64, 0, 1);
      __builtin_prefetch(Xn + (size_t)tid * 64 + 32, 0, 1);
    }

#pragma unroll
    for (int cch = 0; cch < 8; ++cch) {
      int k0 = cch * 32;
      v16h a0 = load_afrag(hsh, 256, mPair * 16, k0, lane);
      v16h a1 = load_afrag(hsh, 256, mPair * 16 + 16, k0, lane);
#pragma unroll
      for (int g = 0; g < 4; ++g) {
        v16h b = (g < 2)
                     ? load_bfrag(wldsO, 256, g * 256 + ht * 16, k0, lane)   // LDS
                     : load_bfrag(WhhO, 256, g * 256 + ht * 16, k0, lane);   // L2
        acc[g][0] = WMMA16(a0, b, acc[g][0]);
        acc[g][1] = WMMA16(a1, b, acc[g][1]);
      }
    }
    __syncthreads();  // all LDS h reads done

    int j = ht * 16 + nl;
#pragma unroll
    for (int mi = 0; mi < 2; ++mi) {
#pragma unroll
      for (int e = 0; e < 8; ++e) {
        float iv = acc[0][mi][e], fv = acc[1][mi][e];
        float gv = acc[2][mi][e], ov = acc[3][mi][e];
        float cn = sig_(fv) * c[mi][e] + sig_(iv) * tanh_(gv);
        float hn = sig_(ov) * tanh_(cn);
        c[mi][e] = cn;
        int m = (mPair + mi) * 16 + 8 * laneHi + e;
        half_t hb = (half_t)hn;
        hsh[m * 256 + j] = hb;
        Hout[((size_t)t * 64 + m) * 256 + j] = hb;
      }
    }
    __syncthreads();  // h_new visible for next step
  }
}

// ---------------------------------------------------------------------------
// Persistent encoder scan: per step (t>0) h_z LSTM (K = 32[z_prev] + 256[h_z]),
// enc = tanh([h_s[t], h_z] @ Wd^T + bd) (K=512), then mean/logvar/z (K=256).
// LDS: h_z 32KB, z_prev 4KB, enc 32KB, Wzhh rows [0,448) cached (224KB).
// ---------------------------------------------------------------------------
__global__ void __launch_bounds__(1024) k_enc_scan(
    const half_t* __restrict__ Hs, const float* __restrict__ eps,
    const half_t* __restrict__ Wzih, const half_t* __restrict__ Wzhh,
    const float* __restrict__ bzih, const float* __restrict__ bzhh,
    const half_t* __restrict__ Wd, const float* __restrict__ bd,
    const half_t* __restrict__ Wm, const float* __restrict__ bm,
    const half_t* __restrict__ Wl, const float* __restrict__ bl,
    float* __restrict__ meanO, float* __restrict__ logvO, float* __restrict__ zO,
    half_t* __restrict__ Zh, int T) {
  extern __shared__ v4u dyn_smem_v4[];               // typed -> stays AS(3)
  half_t* hz    = (half_t*)dyn_smem_v4;
  half_t* zprev = hz + 64 * 256;
  half_t* encs  = zprev + 64 * 32;
  half_t* wzl   = encs + 64 * 256;  // Wzhh rows [0,448): 224KB

  int tid = threadIdx.x, lane = tid & 31, w = tid >> 5;
  int laneHi = lane >> 4, nl = lane & 15;
  int ht = w >> 1, mPair = (w & 1) * 2;

  for (int i = tid; i < 64 * 256; i += 1024) { hz[i] = (half_t)0.f; encs[i] = (half_t)0.f; }
  for (int i = tid; i < 64 * 32; i += 1024) zprev[i] = (half_t)0.f;
  {  // cooperative prefill of the LDS weight cache (448 rows x 256 f16)
    const v4u* src = (const v4u*)Wzhh;
    v4u* dst = (v4u*)wzl;
    for (int i = tid; i < (448 * 256) / 8; i += 1024) dst[i] = src[i];
  }
  v8f cz[2];
  cz[0] = v8f_zero();
  cz[1] = v8f_zero();
  __syncthreads();

  for (int t = 0; t < T; ++t) {
    int zo = opaque_zero_();
    const half_t* WzihO = Wzih + zo;
    const half_t* WzhhO = Wzhh + zo;
    const half_t* wzlO  = wzl + zo;
    const half_t* WdO   = Wd + zo;
    const half_t* WmO   = Wm + zo;
    const half_t* WlO   = Wl + zo;

    // ---- Phase L: h_z, c_z = lstm_cell(z_prev, h_z, c_z)  (skipped at t==0)
    if (t > 0) {
      v8f acc[4][2];
#pragma unroll
      for (int g = 0; g < 4; ++g) {
        int n = g * 256 + ht * 16 + nl;
        float bz = bzih[n] + bzhh[n];
        v8f a;
#pragma unroll
        for (int e = 0; e < 8; ++e) a[e] = bz;
        acc[g][0] = a;
        acc[g][1] = a;
      }
      {  // K chunk from z_prev (K=32)
        v16h a0 = load_afrag(zprev, 32, mPair * 16, 0, lane);
        v16h a1 = load_afrag(zprev, 32, mPair * 16 + 16, 0, lane);
#pragma unroll
        for (int g = 0; g < 4; ++g) {
          v16h b = load_bfrag(WzihO, 32, g * 256 + ht * 16, 0, lane);
          acc[g][0] = WMMA16(a0, b, acc[g][0]);
          acc[g][1] = WMMA16(a1, b, acc[g][1]);
        }
      }
#pragma unroll
      for (int cch = 0; cch < 8; ++cch) {  // K chunks from h_z (K=256)
        int k0 = cch * 32;
        v16h a0 = load_afrag(hz, 256, mPair * 16, k0, lane);
        v16h a1 = load_afrag(hz, 256, mPair * 16 + 16, k0, lane);
#pragma unroll
        for (int g = 0; g < 4; ++g) {
          int nb = g * 256 + ht * 16;
          v16h b = (nb < 448) ? load_bfrag(wzlO, 256, nb, k0, lane)   // LDS cache
                              : load_bfrag(WzhhO, 256, nb, k0, lane); // L2
          acc[g][0] = WMMA16(a0, b, acc[g][0]);
          acc[g][1] = WMMA16(a1, b, acc[g][1]);
        }
      }
      __syncthreads();
      int j = ht * 16 + nl;
#pragma unroll
      for (int mi = 0; mi < 2; ++mi) {
#pragma unroll
        for (int e = 0; e < 8; ++e) {
          float iv = acc[0][mi][e], fv = acc[1][mi][e];
          float gv = acc[2][mi][e], ov = acc[3][mi][e];
          float cn = sig_(fv) * cz[mi][e] + sig_(iv) * tanh_(gv);
          float hn = sig_(ov) * tanh_(cn);
          cz[mi][e] = cn;
          int m = (mPair + mi) * 16 + 8 * laneHi + e;
          hz[m * 256 + j] = (half_t)hn;
        }
      }
      __syncthreads();
    }

    // ---- Phase E: enc = tanh([h_s[t], h_z] @ Wd^T + bd), 64x256, K=512
    {
      v8f aE[2];
      aE[0] = v8f_zero();
      aE[1] = v8f_zero();
      const half_t* HsT = Hs + (size_t)t * 64 * 256;
#pragma unroll
      for (int cch = 0; cch < 16; ++cch) {
        int k0 = cch * 32;
        v16h a0, a1;
        if (cch < 8) {
          a0 = load_afrag(HsT, 256, mPair * 16, k0, lane);
          a1 = load_afrag(HsT, 256, mPair * 16 + 16, k0, lane);
        } else {
          a0 = load_afrag(hz, 256, mPair * 16, k0 - 256, lane);
          a1 = load_afrag(hz, 256, mPair * 16 + 16, k0 - 256, lane);
        }
        v16h b = load_bfrag(WdO, 512, ht * 16, k0, lane);
        aE[0] = WMMA16(a0, b, aE[0]);
        aE[1] = WMMA16(a1, b, aE[1]);
      }
      int n = ht * 16 + nl;
      float bdv = bd[n];
#pragma unroll
      for (int mi = 0; mi < 2; ++mi) {
#pragma unroll
        for (int e = 0; e < 8; ++e) {
          int m = (mPair + mi) * 16 + 8 * laneHi + e;
          encs[m * 256 + n] = (half_t)tanh_(aE[mi][e] + bdv);
        }
      }
    }
    __syncthreads();

    // ---- Phase M: mean/logvar/z (64x32 each, K=256); waves 0..7 only.
    if (w < 8) {
      int mT = w & 3, nT = w >> 2;
      v8f am = v8f_zero(), al = v8f_zero();
#pragma unroll
      for (int cch = 0; cch < 8; ++cch) {
        int k0 = cch * 32;
        v16h a  = load_afrag(encs, 256, mT * 16, k0, lane);
        v16h bM = load_bfrag(WmO, 256, nT * 16, k0, lane);
        v16h bL = load_bfrag(WlO, 256, nT * 16, k0, lane);
        am = WMMA16(a, bM, am);
        al = WMMA16(a, bL, al);
      }
      int n = nT * 16 + nl;
      float bmv = bm[n], blv = bl[n];
#pragma unroll
      for (int e = 0; e < 8; ++e) {
        int m = mT * 16 + 8 * laneHi + e;
        size_t idx = ((size_t)t * 64 + m) * 32 + n;
        float mv = am[e] + bmv;
        float lv = al[e] + blv;
        float zv = eps[idx] * __expf(0.5f * lv) + mv;
        meanO[idx] = mv;
        logvO[idx] = lv;
        zO[idx]    = zv;
        half_t zb = (half_t)zv;
        zprev[m * 32 + n] = zb;
        Zh[idx] = zb;
      }
    }
    __syncthreads();  // z_prev ready for step t+1
  }
}

// ---------------------------------------------------------------------------
// Host launcher
// ---------------------------------------------------------------------------
static inline int imin_(int a, int b) { return a < b ? a : b; }

extern "C" void kernel_launch(void* const* d_in, const int* in_sizes, int n_in,
                              void* d_out, int out_size, void* d_ws, size_t ws_size,
                              hipStream_t stream) {
  (void)in_sizes; (void)n_in; (void)out_size; (void)ws_size;

  const float* s       = (const float*)d_in[0];
  const float* eps     = (const float*)d_in[1];
  const float* Ws_ih   = (const float*)d_in[2];
  const float* Ws_hh   = (const float*)d_in[3];
  const float* bs_ih   = (const float*)d_in[4];
  const float* bs_hh   = (const float*)d_in[5];
  const float* Wz_ih   = (const float*)d_in[6];
  const float* Wz_hh   = (const float*)d_in[7];
  const float* bz_ih   = (const float*)d_in[8];
  const float* bz_hh   = (const float*)d_in[9];
  const float* Wd      = (const float*)d_in[10];
  const float* bd      = (const float*)d_in[11];
  const float* Wm      = (const float*)d_in[12];
  const float* bm      = (const float*)d_in[13];
  const float* Wl      = (const float*)d_in[14];
  const float* bl      = (const float*)d_in[15];
  const float* Wdec_ih = (const float*)d_in[16];
  const float* Wdec_hh = (const float*)d_in[17];
  const float* bdec_ih = (const float*)d_in[18];
  const float* bdec_hh = (const float*)d_in[19];
  const float* Wout    = (const float*)d_in[20];
  const float* bout    = (const float*)d_in[21];

  constexpr int T = 512, Bsz = 64, D = 513, H = 256, Z = 32;
  constexpr int TB = T * Bsz;          // 32768
  constexpr int G = 4 * H;             // 1024
  constexpr int Dp = 544;              // D padded to mult of 32 (K dim)
  constexpr int Doutp = 528;           // D padded to mult of 16 (N dim)

  char* ws = (char*)d_ws;
  size_t off = 0;
  auto alloc = [&](size_t bytes) -> char* {
    char* p = ws + off;
    off = (off + bytes + 255) & ~(size_t)255;
    return p;
  };
  float*  X      = (float*)alloc((size_t)TB * G * 4);     // shared by Xs / Xdec (tiled)
  half_t* s_p    = (half_t*)alloc((size_t)TB * Dp * 2);
  half_t* hs_p   = (half_t*)alloc((size_t)TB * H * 2);
  half_t* hd_p   = (half_t*)alloc((size_t)TB * H * 2);
  half_t* z_p    = (half_t*)alloc((size_t)TB * Z * 2);
  half_t* Wsih_p = (half_t*)alloc((size_t)G * Dp * 2);
  half_t* Wshh_p = (half_t*)alloc((size_t)G * H * 2);
  half_t* Wzih_p = (half_t*)alloc((size_t)G * Z * 2);
  half_t* Wzhh_p = (half_t*)alloc((size_t)G * H * 2);
  half_t* Wd_p   = (half_t*)alloc((size_t)H * 2 * H * 2);
  half_t* Wm_p   = (half_t*)alloc((size_t)Z * H * 2);
  half_t* Wl_p   = (half_t*)alloc((size_t)Z * H * 2);
  half_t* Wdih_p = (half_t*)alloc((size_t)G * Z * 2);
  half_t* Wdhh_p = (half_t*)alloc((size_t)G * H * 2);
  half_t* Wout_p = (half_t*)alloc((size_t)Doutp * H * 2);

  float* xOut  = (float*)d_out;                 // (T,B,513)
  float* meanO = xOut + (size_t)TB * D;         // (T,B,32)
  float* logvO = meanO + (size_t)TB * Z;
  float* zO    = logvO + (size_t)TB * Z;

  auto pack = [&](const float* src, half_t* dst, int R, int C, int Rp, int Cp) {
    int total = Rp * Cp;
    int blocks = imin_((total + 255) / 256, 8192);
    k_pack<<<dim3(blocks), dim3(256), 0, stream>>>(src, dst, R, C, Rp, Cp);
  };

  pack(s, s_p, TB, D, TB, Dp);
  pack(Ws_ih, Wsih_p, G, D, G, Dp);
  pack(Ws_hh, Wshh_p, G, H, G, H);
  pack(Wz_ih, Wzih_p, G, Z, G, Z);
  pack(Wz_hh, Wzhh_p, G, H, G, H);
  pack(Wd, Wd_p, H, 2 * H, H, 2 * H);
  pack(Wm, Wm_p, Z, H, Z, H);
  pack(Wl, Wl_p, Z, H, Z, H);
  pack(Wdec_ih, Wdih_p, G, Z, G, Z);
  pack(Wdec_hh, Wdhh_p, G, H, G, H);
  pack(Wout, Wout_p, D, H, Doutp, H);

  // 1) Xs = s @ Ws_ih^T + bs_ih + bs_hh  (32768 x 1024, K=544, tiled out)
  k_gemm<Dp, true, false, false><<<dim3(G / 256, TB / 64), dim3(256), 0, stream>>>(
      s_p, Wsih_p, bs_ih, bs_hh, X, G, G, G);

  // 2) backward scan -> h_s (f16);  LDS = h(32KB) + Whh cache(256KB)
  size_t scanSmem = (size_t)64 * 256 * 2 + (size_t)512 * 256 * 2;
  k_lstm_scan<<<dim3(1), dim3(1024), scanSmem, stream>>>(X, Wshh_p, hs_p, T, 1);

  // 3) encoder scan;  LDS = hz+zprev+enc(68KB) + Wzhh cache(224KB)
  size_t encSmem =
      (size_t)64 * 256 * 2 + 64 * 32 * 2 + 64 * 256 * 2 + (size_t)448 * 256 * 2;
  k_enc_scan<<<dim3(1), dim3(1024), encSmem, stream>>>(
      hs_p, eps, Wzih_p, Wzhh_p, bz_ih, bz_hh, Wd_p, bd, Wm_p, bm, Wl_p, bl,
      meanO, logvO, zO, z_p, T);

  // 4) Xdec = z @ Wdec_ih^T + bdec_ih + bdec_hh  (K=32, tiled out, reuse X)
  k_gemm<Z, true, false, false><<<dim3(G / 256, TB / 64), dim3(256), 0, stream>>>(
      z_p, Wdih_p, bdec_ih, bdec_hh, X, G, G, G);

  // 5) forward scan -> h_d (f16)
  k_lstm_scan<<<dim3(1), dim3(1024), scanSmem, stream>>>(X, Wdhh_p, hd_p, T, 0);

  // 6) x = exp(h_d @ Wout^T + bout)  (N=513, padded to 528, edge-masked)
  k_gemm<H, false, true, true><<<dim3((Doutp + 255) / 256, TB / 64), dim3(256), 0, stream>>>(
      hd_p, Wout_p, bout, nullptr, xOut, Doutp, D, D);
}
